// CustomLSTM_37692632990156
// MI455X (gfx1250) — matrix operand order repaired and checked by reading, not compiled, not run on hardware
//
#include <hip/hip_runtime.h>
#include <hip/hip_bf16.h>

// ---------------------------------------------------------------------------
// LSTM for MI455X (gfx1250, wave32, WMMA + async-to-LDS staging).
// bf16 WMMA (v_wmma_f32_16x16x32_bf16) with f32 accumulation.
// Staging uses GLOBAL_LOAD_ASYNC_TO_LDS_B128 (ASYNCcnt) with LDS double-buffer.
// ---------------------------------------------------------------------------

typedef __attribute__((ext_vector_type(16))) __bf16 v16bf;
typedef __attribute__((ext_vector_type(8)))  __bf16 bf16x8;
typedef __attribute__((ext_vector_type(4)))  __bf16 bf16x4;
typedef __attribute__((ext_vector_type(8)))  float  v8f;
typedef __attribute__((ext_vector_type(4)))  float  f32x4;

constexpr int Bb = 64;         // batch
constexpr int Tt = 512;        // time steps
constexpr int Dd = 1024;       // input dim
constexpr int Hh = 1024;       // hidden dim
constexpr int G4 = 4 * Hh;     // 4096 fused gate columns (i,f,g,o concat)
constexpr int BT = Bb * Tt;    // 32768 rows of x_proj
constexpr int KC  = 64;        // K chunk = 2 WMMA K-steps
constexpr int KCP = 72;        // padded LDS row (144B: 16B-aligned, bank-spread)

// --------------------------- CDNA5 async copy ------------------------------
// GLOBAL_LOAD_ASYNC_TO_LDS_B128: per-lane 16B global->LDS, tracked by ASYNCcnt.
__device__ __forceinline__ void async_copy16(const __bf16* lds_dst, const __bf16* gsrc) {
    unsigned lds_off = (unsigned)(uintptr_t)lds_dst;      // LDS addr = addr[31:0]
    unsigned long long ga = (unsigned long long)(uintptr_t)gsrc;
    asm volatile("global_load_async_to_lds_b128 %0, %1, off"
                 :: "v"(lds_off), "v"(ga)
                 : "memory");
}
__device__ __forceinline__ void wait_async0() {
    asm volatile("s_wait_asynccnt 0x0" ::: "memory");
}

// ------------------------- WMMA fragment helpers ---------------------------
__device__ __forceinline__ v16bf mk_frag(bf16x8 lo, bf16x8 hi) {
    v16bf r;
#pragma unroll
    for (int i = 0; i < 8; ++i) { r[i] = lo[i]; r[i + 8] = hi[i]; }
    return r;
}
// A (16Mx32K): lane<16 -> K{0..7},{16..23}; lane>=16 -> K{8..15},{24..31}
__device__ __forceinline__ v16bf load_fragA(const __bf16* row_base, int lh) {
    const __bf16* p = row_base + lh * 8;
    return mk_frag(*(const bf16x8*)p, *(const bf16x8*)(p + 16));
}
// B (32Kx16N stored [N][K]): lane<16 -> K{0..7},{16..23}; lane>=16 -> K{8..15},{24..31}
__device__ __forceinline__ v16bf load_fragB(const __bf16* row_base, int lh) {
    const __bf16* p = row_base + lh * 8;
    return mk_frag(*(const bf16x8*)p, *(const bf16x8*)(p + 16));
}

__device__ __forceinline__ float sigmoidf_(float x) {
    return 1.0f / (1.0f + __expf(-x));
}

// ---------------------------------------------------------------------------
// Packing / conversion kernels
// ---------------------------------------------------------------------------
__global__ __launch_bounds__(256) void pack4_bf16_kernel(
    const float* __restrict__ s0, const float* __restrict__ s1,
    const float* __restrict__ s2, const float* __restrict__ s3,
    __bf16* __restrict__ dst, int per)
{
    size_t i = (size_t)blockIdx.x * 256 + threadIdx.x;
    int g = (int)(i / per);
    size_t r = i - (size_t)g * per;
    const float* s = (g == 0) ? s0 : (g == 1) ? s1 : (g == 2) ? s2 : s3;
    dst[i] = (__bf16)s[r];
}

__global__ __launch_bounds__(256) void pack4_f32_kernel(
    const float* __restrict__ s0, const float* __restrict__ s1,
    const float* __restrict__ s2, const float* __restrict__ s3,
    float* __restrict__ dst, int per)
{
    size_t i = (size_t)blockIdx.x * 256 + threadIdx.x;
    int g = (int)(i / per);
    size_t r = i - (size_t)g * per;
    const float* s = (g == 0) ? s0 : (g == 1) ? s1 : (g == 2) ? s2 : s3;
    dst[i] = s[r];
}

// f32 -> bf16 bulk convert (4 elems/thread)
__global__ __launch_bounds__(256) void cvt_bf16_kernel(
    const float* __restrict__ s, __bf16* __restrict__ d)
{
    size_t i = ((size_t)blockIdx.x * 256 + threadIdx.x) * 4;
    f32x4 v = *(const f32x4*)(s + i);
    bf16x4 o;
    o[0] = (__bf16)v[0]; o[1] = (__bf16)v[1];
    o[2] = (__bf16)v[2]; o[3] = (__bf16)v[3];
    *(bf16x4*)(d + i) = o;
}

__global__ __launch_bounds__(256) void init_state_kernel(
    const float* __restrict__ h0, const float* __restrict__ c0,
    __bf16* __restrict__ hb, float* __restrict__ cb)
{
    int i = blockIdx.x * 256 + threadIdx.x;
    hb[i] = (__bf16)h0[i];
    cb[i] = c0[i];
}

// h_T from h_seq[:, T-1, :] (full f32 precision), c_T from cbuf.
__global__ __launch_bounds__(256) void finalize_kernel(
    const float* __restrict__ h_seq, const float* __restrict__ c,
    float* __restrict__ out_tail)
{
    int i = blockIdx.x * 256 + threadIdx.x;
    int m = i >> 10, n = i & (Hh - 1);
    out_tail[i] = h_seq[((size_t)m * Tt + (Tt - 1)) * Hh + n];
    out_tail[Bb * Hh + i] = c[i];
}

// ---------------------------------------------------------------------------
// GEMM 1: x_proj[BT, 4H] = Xb[BT, D](bf16) @ Wi[4H, D]^T + bias
// Block tile 128x128, 8 waves, wave = 64x32 (4x2 WMMA tiles), KC=64,
// async-to-LDS double-buffered staging.
// ---------------------------------------------------------------------------
__global__ __launch_bounds__(256) void xproj_gemm_kernel(
    const __bf16* __restrict__ X, const __bf16* __restrict__ W,
    const float* __restrict__ bias, float* __restrict__ Y)
{
    __shared__ __bf16 As[2][128 * KCP];
    __shared__ __bf16 Bs[2][128 * KCP];

    const int tid  = threadIdx.x;
    const int wave = tid >> 5;
    const int lane = tid & 31;
    const int lr   = lane & 15;
    const int lh   = lane >> 4;
    const int m0 = blockIdx.y * 128;
    const int n0 = blockIdx.x * 128;
    const int wm = (wave >> 2) * 64;
    const int wn = (wave & 3) * 32;

    // Stage one KC=64 chunk of A and B into LDS buffer `buf` (async).
    auto stage = [&](int buf, int k0) {
#pragma unroll
        for (int c = 0; c < 4; ++c) {            // A: 128 rows x 8 chunks
            int lin = tid + c * 256;
            int row = lin >> 3, ck = lin & 7;
            async_copy16(&As[buf][row * KCP + ck * 8],
                         X + (size_t)(m0 + row) * Dd + k0 + ck * 8);
        }
#pragma unroll
        for (int c = 0; c < 4; ++c) {            // B: 128 rows x 8 chunks
            int lin = tid + c * 256;
            int row = lin >> 3, ck = lin & 7;
            async_copy16(&Bs[buf][row * KCP + ck * 8],
                         W + (size_t)(n0 + row) * Dd + k0 + ck * 8);
        }
    };

    const v8f vzero = {0.f, 0.f, 0.f, 0.f, 0.f, 0.f, 0.f, 0.f};
    v8f acc[4][2];
#pragma unroll
    for (int i = 0; i < 4; ++i)
#pragma unroll
        for (int j = 0; j < 2; ++j) acc[i][j] = vzero;

    stage(0, 0);
    wait_async0();
    __syncthreads();

    for (int k = 0; k < Dd / KC; ++k) {
        const int cur = k & 1;
        if (k + 1 < Dd / KC) stage(cur ^ 1, (k + 1) * KC);

#pragma unroll
        for (int ks = 0; ks < 2; ++ks) {         // two WMMA K-steps per chunk
            const int ko = ks * 32;
            v16bf af[4], bfm[2];
#pragma unroll
            for (int i = 0; i < 4; ++i)
                af[i] = load_fragA(&As[cur][(wm + i * 16 + lr) * KCP + ko], lh);
#pragma unroll
            for (int j = 0; j < 2; ++j)
                bfm[j] = load_fragB(&Bs[cur][(wn + j * 16 + lr) * KCP + ko], lh);
#pragma unroll
            for (int i = 0; i < 4; ++i)
#pragma unroll
                for (int j = 0; j < 2; ++j)
                    acc[i][j] = __builtin_amdgcn_wmma_f32_16x16x32_bf16(
                        false, af[i], false, bfm[j], (short)0, acc[i][j], false, false);
        }
        wait_async0();
        __syncthreads();
    }

    // Epilogue: C tile VGPR v, lane -> row = v + 8*lh, col = lr
#pragma unroll
    for (int j = 0; j < 2; ++j) {
        int col = n0 + wn + j * 16 + lr;
        float bia = bias[col];
#pragma unroll
        for (int i = 0; i < 4; ++i) {
#pragma unroll
            for (int v = 0; v < 8; ++v) {
                int row = m0 + wm + i * 16 + lh * 8 + v;
                Y[(size_t)row * G4 + col] = acc[i][j][v] + bia;
            }
        }
    }
}

// ---------------------------------------------------------------------------
// LSTM step t: gates = x_proj[:,t,:] + h @ Wh^T; elementwise update.
// Grid = H/16 = 64 WGs; WG = 64 batch x 16 units x 4 gates.
// 8 waves: gate g = w&3, M-half mh = w>>2. h state kept in bf16.
// ---------------------------------------------------------------------------
__global__ __launch_bounds__(256) void lstm_step_kernel(
    const __bf16* __restrict__ Wh, const float* __restrict__ xproj,
    const __bf16* __restrict__ h_in, __bf16* __restrict__ h_out,
    float* __restrict__ cbuf, float* __restrict__ h_seq, int t)
{
    __shared__ __bf16 Hs[2][64 * KCP];         // A: h (64 x 64K) bf16
    __shared__ __bf16 Bs[2][64 * KCP];         // B: 4 gates x 16 unit rows
    __shared__ float  Gs[4][64][16];           // gate pre-activations

    const int tid  = threadIdx.x;
    const int wave = tid >> 5;
    const int lane = tid & 31;
    const int lr   = lane & 15;
    const int lh   = lane >> 4;
    const int g    = wave & 3;
    const int mh   = wave >> 2;
    const int n0   = blockIdx.x * 16;

    auto stage = [&](int buf, int k0) {
#pragma unroll
        for (int c = 0; c < 2; ++c) {          // A: 64 rows x 8 chunks
            int lin = tid + c * 256;
            int row = lin >> 3, ck = lin & 7;
            async_copy16(&Hs[buf][row * KCP + ck * 8],
                         h_in + (size_t)row * Hh + k0 + ck * 8);
        }
#pragma unroll
        for (int c = 0; c < 2; ++c) {          // B: 64 rows (g*16+j) x 8 chunks
            int lin = tid + c * 256;
            int row = lin >> 3, ck = lin & 7;
            int gg = row >> 4, j = row & 15;
            async_copy16(&Bs[buf][row * KCP + ck * 8],
                         Wh + (size_t)(gg * Hh + n0 + j) * Hh + k0 + ck * 8);
        }
    };

    const v8f vzero = {0.f, 0.f, 0.f, 0.f, 0.f, 0.f, 0.f, 0.f};
    v8f acc[2] = {vzero, vzero};

    stage(0, 0);
    wait_async0();
    __syncthreads();

    for (int k = 0; k < Hh / KC; ++k) {
        const int cur = k & 1;
        if (k + 1 < Hh / KC) stage(cur ^ 1, (k + 1) * KC);

#pragma unroll
        for (int ks = 0; ks < 2; ++ks) {
            const int ko = ks * 32;
            v16bf bfrag = load_fragB(&Bs[cur][(g * 16 + lr) * KCP + ko], lh);
#pragma unroll
            for (int i = 0; i < 2; ++i) {
                v16bf a = load_fragA(&Hs[cur][(mh * 32 + i * 16 + lr) * KCP + ko], lh);
                acc[i] = __builtin_amdgcn_wmma_f32_16x16x32_bf16(
                    false, a, false, bfrag, (short)0, acc[i], false, false);
            }
        }
        wait_async0();
        __syncthreads();
    }

    // Spill gate accumulators for cross-gate combine.
#pragma unroll
    for (int i = 0; i < 2; ++i)
#pragma unroll
        for (int v = 0; v < 8; ++v)
            Gs[g][mh * 32 + i * 16 + lh * 8 + v][lr] = acc[i][v];
    __syncthreads();

    // Elementwise LSTM update: 64x16 = 1024 elems, 4 per thread.
#pragma unroll
    for (int c = 0; c < 4; ++c) {
        int e = tid + c * 256;
        int m = e >> 4, j = e & 15;
        int n = n0 + j;
        const float* xp = xproj + ((size_t)m * Tt + t) * G4;
        float gi = sigmoidf_(Gs[0][m][j] + xp[0 * Hh + n]);
        float gf = sigmoidf_(Gs[1][m][j] + xp[1 * Hh + n]);
        float gc = tanhf   (Gs[2][m][j] + xp[2 * Hh + n]);
        float go = sigmoidf_(Gs[3][m][j] + xp[3 * Hh + n]);
        float cc = gf * cbuf[m * Hh + n] + gi * gc;
        cbuf[m * Hh + n] = cc;
        float hh = go * tanhf(cc);
        h_out[m * Hh + n] = (__bf16)hh;        // bf16 state for next step's GEMM
        h_seq[((size_t)m * Tt + t) * Hh + n] = hh;  // full-precision output
    }
}

// ---------------------------------------------------------------------------
// Host-side orchestration
// ---------------------------------------------------------------------------
extern "C" void kernel_launch(void* const* d_in, const int* in_sizes, int n_in,
                              void* d_out, int out_size, void* d_ws, size_t ws_size,
                              hipStream_t stream) {
    const float* X    = (const float*)d_in[0];   // [B,T,D]
    const float* h0   = (const float*)d_in[1];
    const float* c0   = (const float*)d_in[2];
    const float* W_ii = (const float*)d_in[3];
    const float* W_hi = (const float*)d_in[4];
    const float* b_i  = (const float*)d_in[5];
    const float* W_if = (const float*)d_in[6];
    const float* W_hf = (const float*)d_in[7];
    const float* b_f  = (const float*)d_in[8];
    const float* W_ig = (const float*)d_in[9];
    const float* W_hg = (const float*)d_in[10];
    const float* b_g  = (const float*)d_in[11];
    const float* W_io = (const float*)d_in[12];
    const float* W_ho = (const float*)d_in[13];
    const float* b_o  = (const float*)d_in[14];

    float* out = (float*)d_out;                  // h_seq [B,T,H] then h_T, c_T
    char*  ws  = (char*)d_ws;

    auto align256 = [](size_t x) { return (x + 255) & ~(size_t)255; };
    size_t off = 0;
    __bf16* Wi   = (__bf16*)(ws + off); off = align256(off + (size_t)G4 * Dd * sizeof(__bf16));
    __bf16* Wh   = (__bf16*)(ws + off); off = align256(off + (size_t)G4 * Hh * sizeof(__bf16));
    float*  bias = (float*) (ws + off); off = align256(off + (size_t)G4 * sizeof(float));
    __bf16* hb0  = (__bf16*)(ws + off); off = align256(off + (size_t)Bb * Hh * sizeof(__bf16));
    __bf16* hb1  = (__bf16*)(ws + off); off = align256(off + (size_t)Bb * Hh * sizeof(__bf16));
    float*  cbuf = (float*) (ws + off); off = align256(off + (size_t)Bb * Hh * sizeof(float));
    __bf16* Xb   = (__bf16*)(ws + off); off = align256(off + (size_t)BT * Dd * sizeof(__bf16));
    float*  xproj= (float*) (ws + off); // [BT, 4H] f32 (512 MB)

    // 1. Pack weights / biases (gate order i,f,g,o) and convert X to bf16.
    {
        int per = Hh * Dd;
        pack4_bf16_kernel<<<(4 * per) / 256, 256, 0, stream>>>(W_ii, W_if, W_ig, W_io, Wi, per);
    }
    {
        int per = Hh * Hh;
        pack4_bf16_kernel<<<(4 * per) / 256, 256, 0, stream>>>(W_hi, W_hf, W_hg, W_ho, Wh, per);
    }
    pack4_f32_kernel<<<G4 / 256, 256, 0, stream>>>(b_i, b_f, b_g, b_o, bias, Hh);
    init_state_kernel<<<(Bb * Hh) / 256, 256, 0, stream>>>(h0, c0, hb0, cbuf);
    cvt_bf16_kernel<<<((size_t)BT * Dd / 4) / 256, 256, 0, stream>>>(X, Xb);

    // 2. Big input-projection GEMM (async-to-LDS + WMMA).
    xproj_gemm_kernel<<<dim3(G4 / 128, BT / 128), 256, 0, stream>>>(Xb, Wi, bias, xproj);

    // 3. Sequential recurrence: 512 step kernels on the same stream.
    __bf16* hbufs[2] = {hb0, hb1};
    for (int t = 0; t < Tt; ++t) {
        lstm_step_kernel<<<Hh / 16, 256, 0, stream>>>(
            Wh, xproj, hbufs[t & 1], hbufs[(t + 1) & 1], cbuf, out, t);
    }

    // 4. Final (h_T, c_T) -> output tail (h_T taken from f32 h_seq).
    finalize_kernel<<<(Bb * Hh) / 256, 256, 0, stream>>>(
        out, cbuf, out + (size_t)Bb * Tt * Hh);
}
